// BaseHysteresis_16630113370990
// MI455X (gfx1250) — matrix operand order repaired and checked by reading, not compiled, not run on hardware
//
#include <hip/hip_runtime.h>
#include <stdint.h>

// ---------------------------------------------------------------------------
// Preisach hysteresis for MI455X (gfx1250, wave32).
// Phase 1: mesh-parallel sequential relay scan (transcendental-bound),
//          states streamed to HBM as bf16 ([T][NPAD] row-major).
// Phase 2: m = states @ density as a WMMA GEMV (v_wmma_f32_16x16x32_bf16),
//          fused with the affine epilogue.
// Workspace requirement: 65536 + T_HIST*NPAD*2 bytes  (~165 MB).
// ---------------------------------------------------------------------------

#define T_HIST 4096
#define N_MESH 20100
#define NPAD   20128              // multiple of 32 (629 K-tiles of 32)
#define NT_TILES (NPAD / 32)      // 629
// 1/TEMP * log2(e): sigmoid(x/temp) = rcp(1 + exp2(-x * C_SCALE))
#define C_SCALE 144.26950408889634f

typedef __attribute__((ext_vector_type(16))) __bf16 v16bf;
typedef __attribute__((ext_vector_type(8)))  float  v8f;

__device__ __forceinline__ unsigned short f32_to_bf16(float f) {
    union { float f; uint32_t u; } x; x.f = f;
    uint32_t u = x.u;
    uint32_t r = u + 0x7FFFu + ((u >> 16) & 1u);   // round to nearest even
    return (unsigned short)(r >> 16);
}

// ---------------------------------------------------------------------------
// Phase 0: Sum(density) in f32, and density -> bf16 table (padded with zeros).
// Single block; trivially fast.
// ---------------------------------------------------------------------------
__global__ void hyst_phase0(const float* __restrict__ density,
                            float* __restrict__ sumd,
                            unsigned short* __restrict__ dbf) {
    __shared__ float red[8];
    int tid = threadIdx.x;
    float acc = 0.f;
    for (int i = tid; i < NPAD; i += 256) {
        float d = (i < N_MESH) ? density[i] : 0.f;
        dbf[i] = f32_to_bf16(d);
        acc += d;
    }
    for (int o = 16; o > 0; o >>= 1) acc += __shfl_xor(acc, o, 32);
    if ((tid & 31) == 0) red[tid >> 5] = acc;
    __syncthreads();
    if (tid == 0) {
        float s = 0.f;
        for (int w = 0; w < 8; ++w) s += red[w];
        sumd[0] = s;
    }
}

// ---------------------------------------------------------------------------
// Phase 1: each thread owns one hysteron, scans all T steps sequentially.
// Per step: 1 LDS broadcast load, 1 v_exp, 1 v_rcp, ~6 VALU, 1 b16 store.
// Direction (h_t > h_prev) is lane-uniform, handled branchlessly.
// ---------------------------------------------------------------------------
__global__ void hyst_phase1(const float* __restrict__ h,
                            const float* __restrict__ mesh,   // [N_MESH][2] = (beta, alpha)
                            unsigned short* __restrict__ S) { // [T_HIST][NPAD] bf16
    __shared__ float sh[T_HIST];
    int tid = threadIdx.x;
    int n = blockIdx.x * 256 + tid;
    for (int i = tid; i < T_HIST; i += 256) sh[i] = h[i];
    __syncthreads();
    if (n >= NPAD) return;

    float beta = 0.f, alpha = 0.f;
    if (n < N_MESH) { beta = mesh[2 * n]; alpha = mesh[2 * n + 1]; }
    const float aC = alpha * C_SCALE;
    const float bC = beta  * C_SCALE;

    float state = -1.f;
    float hprev = 0.f;
    unsigned short* outp = S + n;

    for (int t = 0; t < T_HIST; ++t) {
        float ht  = sh[t];
        bool  up  = ht > hprev;
        float htC = ht * C_SCALE;
        // up:   sg = sigmoid((ht-alpha)/T)  -> exp2 arg = aC - htC
        // down: sg = sigmoid((beta-ht)/T)   -> exp2 arg = htC - bC
        float y  = up ? (aC - htC) : (htC - bC);
        float e  = __builtin_amdgcn_exp2f(y);
        float sg = __builtin_amdgcn_rcpf(1.f + e);
        // up:   s' = s*(1-sg) + sg     down: s' = s*(1-sg) - sg
        float sgn = up ? sg : -sg;
        state = __builtin_fmaf(state, 1.f - sg, sgn);
        hprev = ht;
        outp[(size_t)t * NPAD] = f32_to_bf16(state);
    }
}

// ---------------------------------------------------------------------------
// Phase 2: m[t] = sum_n S[t,n] * d[n] via v_wmma_f32_16x16x32_bf16.
// One wave per 16-row t-strip; loops over 629 K-tiles of 32.
// A (16x32 bf16): lane l -> row l&15; VGPR0-3 = k[(l>>4)*8 ..+8),
//                 VGPR4-7 = k[16+(l>>4)*8 ..+8): two contiguous B128 loads.
// B (32x16 bf16): density broadcast across columns: lane l holds
//                 d[(l>>4)*8 ..+8) and d[16+(l>>4)*8 ..+8): two B128 loads.
// C/D: VGPR i, lanes 0-15 -> row i, lanes 16-31 -> row i+8; all cols equal.
// ---------------------------------------------------------------------------
__global__ void hyst_phase2(const unsigned short* __restrict__ S,
                            const unsigned short* __restrict__ dbf,
                            const float* __restrict__ sumd,
                            const float* __restrict__ h,
                            const float* __restrict__ scale,
                            const float* __restrict__ offset,
                            const float* __restrict__ slope,
                            float* __restrict__ out) {
    int gtid = blockIdx.x * blockDim.x + threadIdx.x;
    int wave = gtid >> 5;
    int lane = threadIdx.x & 31;
    int t0 = wave * 16;
    if (t0 >= T_HIST) return;

    int r    = lane & 15;
    int half = lane >> 4;

    const unsigned short* arow = S + (size_t)(t0 + r) * NPAD + half * 8;
    const unsigned short* brow = dbf + half * 8;

    v8f c = {};
    for (int nt = 0; nt < NT_TILES; ++nt) {
        union { v16bf v; uint4 q[2]; } A, B;
        A.q[0] = *(const uint4*)(arow);
        A.q[1] = *(const uint4*)(arow + 16);
        B.q[0] = *(const uint4*)(brow);
        B.q[1] = *(const uint4*)(brow + 16);
        // Prefetch the A stream a few tiles ahead (global_prefetch_b8).
        __builtin_prefetch(arow + 8 * 32, 0, 0);
        c = __builtin_amdgcn_wmma_f32_16x16x32_bf16(
                /*neg_a=*/false, A.v, /*neg_b=*/false, B.v,
                /*c_mod=*/(short)0, c, /*reuse_a=*/false, /*reuse_b=*/false);
        arow += 32;
        brow += 32;
    }

    if (r == 0) {
        float inv = 1.0f / sumd[0];
        float sc = scale[0], of = offset[0], sl = slope[0];
        #pragma unroll
        for (int i = 0; i < 8; ++i) {
            int t = t0 + i + 8 * half;
            out[t] = sc * (c[i] * inv) + of + h[t] * sl;
        }
    }
}

// ---------------------------------------------------------------------------
// Launch: inputs are (h, mesh_points, density, scale, offset, slope).
// Output: 4096 f32.
// ---------------------------------------------------------------------------
extern "C" void kernel_launch(void* const* d_in, const int* in_sizes, int n_in,
                              void* d_out, int out_size, void* d_ws, size_t ws_size,
                              hipStream_t stream) {
    const float* h     = (const float*)d_in[0];
    const float* mesh  = (const float*)d_in[1];
    const float* dens  = (const float*)d_in[2];
    const float* scale = (const float*)d_in[3];
    const float* offs  = (const float*)d_in[4];
    const float* slope = (const float*)d_in[5];
    float* out = (float*)d_out;

    char* ws = (char*)d_ws;
    float*          sumd = (float*)ws;                        // 4 B
    unsigned short* dbf  = (unsigned short*)(ws + 256);       // NPAD*2 = 40256 B
    unsigned short* S    = (unsigned short*)(ws + 65536);     // T_HIST*NPAD*2 B

    hyst_phase0<<<1, 256, 0, stream>>>(dens, sumd, dbf);
    hyst_phase1<<<(NPAD + 255) / 256, 256, 0, stream>>>(h, mesh, S);
    hyst_phase2<<<(T_HIST / 16) / 8, 256, 0, stream>>>(S, dbf, sumd, h,
                                                       scale, offs, slope, out);
}
